// VolumetricModelV2_39960375722301
// MI455X (gfx1250) — matrix operand order repaired and checked by reading, not compiled, not run on hardware
//
#include <hip/hip_runtime.h>
#include <hip/hip_bf16.h>
#include <math.h>

// ---------------------------------------------------------------- constants
#define BB 2
#define FF 32
#define VV 8192
#define HH 128
#define EE 16
#define BONE_FEAT 768

typedef __attribute__((ext_vector_type(16))) _Float16 v16h;
typedef __attribute__((ext_vector_type(8)))  float    v8f;

union AFrag { v16h v; unsigned u[8]; };
union CFrag { v8f  v; float    f[8]; };

__device__ __forceinline__ float gelu_f(float x) {
    return 0.5f * x * (1.0f + erff(x * 0.70710678118654752f));
}
__device__ __forceinline__ float sigm_f(float x) {
    return 1.0f / (1.0f + expf(-x));
}

// Pack WMMA A-fragment from an LDS row of f16 (row = M index of this lane).
// A 16x32, contiguous k order used here (B matches): lanes 0-15 = M, hi-lanes offset K by 16; VGPR j holds
// K pair base j*2.
__device__ __forceinline__ v16h load_a_frag(const _Float16* xrow, int kstep, int hi) {
    AFrag a;
    const unsigned* p = (const unsigned*)xrow;
    int base = kstep * 32 + hi * 16;
#pragma unroll
    for (int j = 0; j < 8; ++j) {
        int koff = j * 2;
        a.u[j] = p[(base + koff) >> 1];
    }
    return a.v;
}

// B 32x16 fragment: lane (n = lane&15) holds column n; lanes 16-31 hold
// K+16..31. Contiguous 16 halfs of row-major W[n][k].
__device__ __forceinline__ v16h load_b_frag(const _Float16* W, int n, int K, int kb) {
    AFrag b;
    const unsigned* p = (const unsigned*)(W + (size_t)n * K + kb);
#pragma unroll
    for (int j = 0; j < 8; ++j) b.u[j] = p[j];
    return b.v;
}

// ---------------------------------------------------------------- small GEMM (VALU)
// Y[r,o] = dot(X[r,:K], W[o*wstride+woff : +K]) + b1[o] + b2[o]
__global__ void k_rowdot(const float* __restrict__ X, const float* __restrict__ W,
                         const float* __restrict__ b1, const float* __restrict__ b2,
                         float* __restrict__ Y, int R, int N, int K, int wstride, int woff) {
    int idx = blockIdx.x * 256 + threadIdx.x;
    if (idx >= R * N) return;
    int r = idx / N, o = idx - r * N;
    const float* wp = W + (size_t)o * wstride + woff;
    const float* xp = X + (size_t)r * K;
    float acc = 0.f;
    for (int k = 0; k < K; ++k) acc += xp[k] * wp[k];
    if (b1) acc += b1[o];
    if (b2) acc += b2[o];
    Y[idx] = acc;
}

// ---------------------------------------------------------------- f32 -> f16 pack
__global__ void k_pack_f16(const float* __restrict__ src, _Float16* __restrict__ dst,
                           int rows, int rowLen, int sstride, int soff) {
    int idx = blockIdx.x * 256 + threadIdx.x;
    if (idx >= rows * rowLen) return;
    int r = idx / rowLen, c = idx - r * rowLen;
    dst[idx] = (_Float16)src[(size_t)r * sstride + soff + c];
}

// ---------------------------------------------------------------- WMMA GEMM
// Y(RxN) = X(RxK) @ Wh(NxK)^T + bias ; R % 64 == 0, K compile-time (128/256),
// N % 16 == 0. K templated so the kstep loops fully unroll and the hoisted
// A-fragments stay in VGPRs (runtime trip count would spill afr[] to scratch).
template <int K>
__global__ __launch_bounds__(128) void k_gemm16(const float* __restrict__ X,
                                                const _Float16* __restrict__ Wh,
                                                const float* __restrict__ bias,
                                                float* __restrict__ Y, int N) {
    constexpr int KSTEPS = K / 32;
    __shared__ __align__(16) _Float16 sX[4][16][K];
    int tid = threadIdx.x, w = tid >> 5, lane = tid & 31;
    int row0 = blockIdx.x * 64 + w * 16;
    for (int r = 0; r < 16; ++r) {
        const float4* xp = (const float4*)(X + (size_t)(row0 + r) * K);
#pragma unroll
        for (int k4 = lane; k4 < (K >> 2); k4 += 32) {
            float4 xv = xp[k4];
            int k = k4 * 4;
            sX[w][r][k + 0] = (_Float16)xv.x;
            sX[w][r][k + 1] = (_Float16)xv.y;
            sX[w][r][k + 2] = (_Float16)xv.z;
            sX[w][r][k + 3] = (_Float16)xv.w;
        }
    }
    int m = lane & 15, hi = lane >> 4;
    v16h afr[KSTEPS];
#pragma unroll
    for (int ks = 0; ks < KSTEPS; ++ks)  // A depends only on kstep: keep in regs
        afr[ks] = load_a_frag(&sX[w][m][0], ks, hi);
    for (int n0 = 0; n0 < N; n0 += 16) {
        CFrag c;
#pragma unroll
        for (int i = 0; i < 8; ++i) c.f[i] = 0.f;
#pragma unroll
        for (int ks = 0; ks < KSTEPS; ++ks) {
            v16h b = load_b_frag(Wh, n0 + m, K, ks * 32 + hi * 16);
            c.v = __builtin_amdgcn_wmma_f32_16x16x32_f16(false, afr[ks], false, b,
                                                         (short)0, c.v, false, false);
        }
        int n = n0 + m;
        float bv = bias ? bias[n] : 0.f;
#pragma unroll
        for (int i = 0; i < 8; ++i)
            Y[(size_t)(row0 + i + 8 * hi) * N + n] = c.f[i] + bv;
    }
}

// ---------------------------------------------------------------- activation/LN
// mode 0: dst = LN(gelu(x))*g+b (+res)   mode 1: dst = gelu(LN(x)*g+b)
// mode 2: dst = LN(x)*g+b                mode 3: dst = gelu(x)
__global__ void k_act(const float* __restrict__ src, float* __restrict__ dst,
                      const float* __restrict__ g, const float* __restrict__ bt,
                      const float* __restrict__ res, int N, int mode) {
    __shared__ float s[512];
    __shared__ float p1[256], p2[256];
    int r = blockIdx.x, t = threadIdx.x;
    float ls = 0.f, lq = 0.f;
    for (int k = t; k < N; k += 256) {
        float x = src[(size_t)r * N + k];
        if (mode == 0 || mode == 3) x = gelu_f(x);
        s[k] = x;
        ls += x; lq += x * x;
    }
    if (mode == 3) {
        __syncthreads();
        for (int k = t; k < N; k += 256) dst[(size_t)r * N + k] = s[k];
        return;
    }
    p1[t] = ls; p2[t] = lq;
    __syncthreads();
    for (int s2 = 128; s2 > 0; s2 >>= 1) {
        if (t < s2) { p1[t] += p1[t + s2]; p2[t] += p2[t + s2]; }
        __syncthreads();
    }
    float mean = p1[0] / (float)N;
    float var = p2[0] / (float)N - mean * mean;
    float rstd = rsqrtf(var + 1e-5f);
    for (int k = t; k < N; k += 256) {
        float y = (s[k] - mean) * rstd * g[k] + bt[k];
        if (mode == 1) y = gelu_f(y);
        if (res) y += res[(size_t)r * N + k];
        dst[(size_t)r * N + k] = y;
    }
}

// ---------------------------------------------------------------- LSTM (one layer)
// xg: (B,F,512) precomputed  x@Wih^T+bih+bhh ; Whh: (512,128) ; hs: (B,F,128)
__global__ __launch_bounds__(1024) void k_lstm(const float* __restrict__ xg,
                                               const float* __restrict__ Whh,
                                               float* __restrict__ hs) {
    __shared__ float sh[2][128], sc[2][128], sg[2][512];
    int t = threadIdx.x;
    int b = t >> 9, gi = t & 511;
    if (t < 256) { sh[t >> 7][t & 127] = 0.f; sc[t >> 7][t & 127] = 0.f; }
    __syncthreads();
    for (int step = 0; step < FF; ++step) {
        float sacc = xg[((size_t)b * FF + step) * 512 + gi];
        const float* wr = Whh + (size_t)gi * 128;
        for (int k = 0; k < 128; ++k) sacc += wr[k] * sh[b][k];
        sg[b][gi] = sacc;
        __syncthreads();
        if (t < 256) {
            int bb = t >> 7, u = t & 127;
            float ig = sigm_f(sg[bb][u]);
            float fg = sigm_f(sg[bb][128 + u]);
            float gg = tanhf(sg[bb][256 + u]);
            float og = sigm_f(sg[bb][384 + u]);
            float c = fg * sc[bb][u] + ig * gg;
            float h = og * tanhf(c);
            sc[bb][u] = c; sh[bb][u] = h;
            hs[((size_t)bb * FF + step) * 128 + u] = h;
        }
        __syncthreads();
    }
}

// ---------------------------------------------------------------- mean over V
__global__ void k_mean(const float* __restrict__ X, float* __restrict__ out) {
    __shared__ float p[256];
    int o = blockIdx.x;            // o = b*128 + c
    int b = o >> 7, c = o & 127;
    int t = threadIdx.x;
    float s = 0.f;
    for (int v = t; v < VV; v += 256) s += X[((size_t)b * VV + v) * 128 + c];
    p[t] = s;
    __syncthreads();
    for (int s2 = 128; s2 > 0; s2 >>= 1) {
        if (t < s2) p[t] += p[t + s2];
        __syncthreads();
    }
    if (t == 0) out[o] = p[0] / (float)VV;
}

// ---------------------------------------------------------------- misc fusion glue
__global__ void k_build_fus(const float* __restrict__ m, const float* __restrict__ rg,
                            const float* __restrict__ sm, float* __restrict__ xcat) {
    int idx = blockIdx.x * 256 + threadIdx.x;
    if (idx >= 64 * 384) return;
    int r = idx / 384, c = idx - r * 384;
    int b = r >> 5;
    float v;
    if (c < 128) v = m[r * 128 + c];
    else if (c < 256) v = rg[b * 128 + (c - 128)];
    else v = sm[b * 128 + (c - 256)];
    xcat[idx] = v;
}

__global__ void k_combine_rv(const float* __restrict__ rvA, const float* __restrict__ vep,
                             const float* __restrict__ b1, float* __restrict__ rv) {
    int idx = blockIdx.x * 256 + threadIdx.x;
    if (idx >= BB * VV * 128) return;
    rv[idx] = rvA[idx] + vep[idx & (VV * 128 - 1)] + b1[idx & 127];
}

// ---------------------------------------------------------------- decoder (WMMA)
// per 16-row tile: h = gelu(rv+dm) -> LN(n1) -> f16 X -> X@W2^T (WMMA f16)
// -> +b2 -> gelu -> LN(n2) -> @W3^T + b3 -> out (3 floats/row)
__global__ __launch_bounds__(128) void k_decoder(const float* __restrict__ rv,
                                                 const float* __restrict__ dm,
                                                 const _Float16* __restrict__ W2h,
                                                 const float* __restrict__ b2,
                                                 const float* __restrict__ g1,
                                                 const float* __restrict__ bt1,
                                                 const float* __restrict__ g2,
                                                 const float* __restrict__ bt2,
                                                 const float* __restrict__ W3,
                                                 const float* __restrict__ b3,
                                                 float* __restrict__ out) {
    __shared__ __align__(16) float s_dm[128];
    __shared__ __align__(16) _Float16 sX[4][16][128];
    __shared__ __align__(16) _Float16 sW2[64 * 128];   // dec_l2 weights, staged once
    __shared__ float sZ[4][16][64];
    __shared__ float s_b2[64], s_g2[64], s_bt2[64];
    __shared__ float s_g1[128], s_bt1[128];
    __shared__ float s_W3[192], s_b3[3];

    int b = blockIdx.z, f = blockIdx.y;
    int tid = threadIdx.x, w = tid >> 5, lane = tid & 31;
    int v0 = blockIdx.x * 64 + w * 16;

    // prefetch this wave's rv tile (global_prefetch_b8) before staging barrier
    {
        const float* pf = rv + ((size_t)b * VV + v0) * 128 + lane * 4;
#pragma unroll
        for (int r = 0; r < 16; ++r) __builtin_prefetch(pf + r * 128, 0, 1);
    }

    if (tid < 128) {
        s_dm[tid] = dm[((size_t)b * FF + f) * 128 + tid];
        s_g1[tid] = g1[tid];
        s_bt1[tid] = bt1[tid];
    }
    if (tid < 64) { s_b2[tid] = b2[tid]; s_g2[tid] = g2[tid]; s_bt2[tid] = bt2[tid]; }
    if (tid < 96) { s_W3[tid] = W3[tid]; s_W3[96 + tid] = W3[96 + tid]; }
    if (tid < 3) s_b3[tid] = b3[tid];
    {   // stage W2 (64x128 f16 = 16KB) into LDS, 8B per thread per iter
        const uint2* src = (const uint2*)W2h;
        uint2* dst = (uint2*)sW2;
        for (int i = tid; i < (64 * 128 / 4); i += 128) dst[i] = src[i];
    }
    __syncthreads();

    // stage 1: h -> gelu -> LN(128) -> f16 into LDS (per-wave tile)
    for (int r = 0; r < 16; ++r) {
        int v = v0 + r;
        const float4* rp = (const float4*)(rv + ((size_t)b * VV + v) * 128);
        float4 xv = rp[lane];
        float4 dv = ((const float4*)s_dm)[lane];
        float gv[4];
        float ls = 0.f, lq = 0.f;
        gv[0] = gelu_f(xv.x + dv.x);
        gv[1] = gelu_f(xv.y + dv.y);
        gv[2] = gelu_f(xv.z + dv.z);
        gv[3] = gelu_f(xv.w + dv.w);
#pragma unroll
        for (int j = 0; j < 4; ++j) { ls += gv[j]; lq += gv[j] * gv[j]; }
#pragma unroll
        for (int off = 16; off >= 1; off >>= 1) {
            ls += __shfl_xor(ls, off);
            lq += __shfl_xor(lq, off);
        }
        float mean = ls * (1.f / 128.f);
        float var = lq * (1.f / 128.f) - mean * mean;
        float rstd = rsqrtf(var + 1e-5f);
#pragma unroll
        for (int j = 0; j < 4; ++j) {
            int k = lane * 4 + j;
            sX[w][r][k] = (_Float16)((gv[j] - mean) * rstd * s_g1[k] + s_bt1[k]);
        }
    }

    // stage 2: Y(16x64) = X(16x128) @ W2^T via WMMA, +b2, gelu -> sZ
    int m = lane & 15, hi = lane >> 4;
    v16h afr[4];
#pragma unroll
    for (int ks = 0; ks < 4; ++ks)      // A frags depend only on kstep: hoist
        afr[ks] = load_a_frag(&sX[w][m][0], ks, hi);
#pragma unroll
    for (int nt = 0; nt < 4; ++nt) {
        CFrag c;
#pragma unroll
        for (int i = 0; i < 8; ++i) c.f[i] = 0.f;
#pragma unroll
        for (int ks = 0; ks < 4; ++ks) {
            v16h bfr = load_b_frag(sW2, nt * 16 + m, 128, ks * 32 + hi * 16);
            c.v = __builtin_amdgcn_wmma_f32_16x16x32_f16(false, afr[ks], false, bfr,
                                                         (short)0, c.v, false, false);
        }
        int n = nt * 16 + m;
#pragma unroll
        for (int i = 0; i < 8; ++i) {
            float z = c.f[i] + s_b2[n];
            sZ[w][i + 8 * hi][n] = gelu_f(z);
        }
    }

    // stage 3: LN(64, n2) -> 64x3 matvec -> out
    for (int r = 0; r < 16; ++r) {
        int v = v0 + r;
        float z0 = sZ[w][r][lane * 2];
        float z1 = sZ[w][r][lane * 2 + 1];
        float ls = z0 + z1, lq = z0 * z0 + z1 * z1;
#pragma unroll
        for (int off = 16; off >= 1; off >>= 1) {
            ls += __shfl_xor(ls, off);
            lq += __shfl_xor(lq, off);
        }
        float mean = ls * (1.f / 64.f);
        float var = lq * (1.f / 64.f) - mean * mean;
        float rstd = rsqrtf(var + 1e-5f);
        float zn0 = (z0 - mean) * rstd * s_g2[lane * 2] + s_bt2[lane * 2];
        float zn1 = (z1 - mean) * rstd * s_g2[lane * 2 + 1] + s_bt2[lane * 2 + 1];
        float o0 = zn0 * s_W3[lane * 2] + zn1 * s_W3[lane * 2 + 1];
        float o1 = zn0 * s_W3[64 + lane * 2] + zn1 * s_W3[64 + lane * 2 + 1];
        float o2 = zn0 * s_W3[128 + lane * 2] + zn1 * s_W3[128 + lane * 2 + 1];
#pragma unroll
        for (int off = 16; off >= 1; off >>= 1) {
            o0 += __shfl_xor(o0, off);
            o1 += __shfl_xor(o1, off);
            o2 += __shfl_xor(o2, off);
        }
        if (lane == 0) {
            size_t base = (((size_t)b * FF + f) * VV + v) * 3;
            out[base + 0] = o0 + s_b3[0];
            out[base + 1] = o1 + s_b3[1];
            out[base + 2] = o2 + s_b3[2];
        }
    }
}

// ================================================================ host
extern "C" void kernel_launch(void* const* d_in, const int* in_sizes, int n_in,
                              void* d_out, int out_size, void* d_ws, size_t ws_size,
                              hipStream_t stream) {
    (void)out_size; (void)ws_size;
    // Detect flattening layout: insertion-order top-level (semantic_vec at [2])
    // vs full sorted-pytree order. Params themselves assumed sorted-key order.
    int pbase;
    const float *bone, *restp, *semv;
    if (in_sizes[2] == 12) {       // [bone, rest_pose, semantic_vec, params...]
        pbase = 3;
        bone = (const float*)d_in[0];
        restp = (const float*)d_in[1];
        semv = (const float*)d_in[2];
    } else {                       // [bone, params..., rest_pose, semantic_vec]
        pbase = 1;
        bone = (const float*)d_in[0];
        restp = (const float*)d_in[n_in - 2];
        semv = (const float*)d_in[n_in - 1];
    }
    auto P = [&](int i) { return (const float*)d_in[pbase + i]; };
    // sorted param leaf indices:
    // 0 dec_l1.W(128x272) 1 dec_l1.b  2 dec_l2.W(64x128) 3 dec_l2.b
    // 4 dec_l3.W(3x64)    5 dec_l3.b  6 dec_ln.g 7 dec_ln.b
    // 8..11 dec_lstm0 (Wih,Whh,bih,bhh)  12..15 dec_lstm1
    // 16 dec_n1.g 17 dec_n1.b 18 dec_n2.g 19 dec_n2.b
    // 20..23 fus0 (W512x384,b,g,bt) 24..27 fus1 (W256x512,...) 28..31 fus2 (W128x256,...)
    // 32 me_lin.W 33 me_lin.b 34 me_ln1.g 35 me_ln1.b 36 me_ln2.g 37 me_ln2.b
    // 38..41 me_lstm0 (Wih 512x768,...) 42..45 me_lstm1
    // 46..49 rest0 (W128x3,b,g,bt) 50..53 rest1 (W256x128) 54..57 rest2 (W256x256)
    // 58..61 rest3 (W128x256) 62..65 sem0 (W128x6) 66..69 sem1 (W128x128) 70 vert_emb

    // ---- workspace carve-up
    float* W = (float*)d_ws;
    size_t off = 0;
    auto allocF = [&](size_t n) { float* p = W + off; off += n; return p; };
    float* A     = allocF((size_t)16384 * 256);   // ping (16MB)
    float* Bbuf  = allocF((size_t)16384 * 256);   // pong (16MB)
    float* restF = allocF((size_t)16384 * 128);   // rest_feat (8MB)
    float* rv    = allocF((size_t)BB * VV * 128); // combined rv (8MB)
    float* xg    = allocF(64 * 512);
    float* hs0   = allocF(64 * 128);
    float* mraw  = allocF(64 * 128);
    float* m1    = allocF(64 * 128);
    float* m2    = allocF(64 * 128);
    float* mfin  = allocF(64 * 128);
    float* semA  = allocF(2 * 128);
    float* semF  = allocF(2 * 128);
    float* restG = allocF(2 * 128);
    float* xcat  = allocF(64 * 384);
    float* f1    = allocF(64 * 512);
    float* f2    = allocF(64 * 256);
    float* fused = allocF(64 * 128);
    float* dh0   = allocF(64 * 128);
    float* dh1   = allocF(64 * 128);
    float* dvec  = allocF(64 * 128);
    float* dm    = allocF(64 * 128);
    _Float16* Hb = (_Float16*)(W + off);
    size_t hoff = 0;
    auto allocH = [&](size_t n) { _Float16* p = Hb + hoff; hoff += n; return p; };
    _Float16* Wrh  = allocH(128 * 128);
    _Float16* W2h  = allocH(64 * 128);
    _Float16* r1Wh = allocH(256 * 128);
    _Float16* r2Wh = allocH(256 * 256);
    _Float16* r3Wh = allocH(128 * 256);
    float* rvA = A;    // reuse: free after rest MLP
    float* vep = Bbuf; // reuse

    auto g1 = [](int n) { return dim3((n + 255) / 256); };

    // ---- pack f16 weights
    k_pack_f16<<<g1(128 * 128), 256, 0, stream>>>(P(0), Wrh, 128, 128, 272, 128);
    k_pack_f16<<<g1(64 * 128), 256, 0, stream>>>(P(2), W2h, 64, 128, 128, 0);
    k_pack_f16<<<g1(256 * 128), 256, 0, stream>>>(P(50), r1Wh, 256, 128, 128, 0);
    k_pack_f16<<<g1(256 * 256), 256, 0, stream>>>(P(54), r2Wh, 256, 256, 256, 0);
    k_pack_f16<<<g1(128 * 256), 256, 0, stream>>>(P(58), r3Wh, 128, 256, 256, 0);

    // ---- motion encoder LSTM (2 layers)
    k_rowdot<<<g1(64 * 512), 256, 0, stream>>>(bone, P(38), P(40), P(41), xg, 64, 512, BONE_FEAT, BONE_FEAT, 0);
    k_lstm<<<1, 1024, 0, stream>>>(xg, P(39), hs0);
    k_rowdot<<<g1(64 * 512), 256, 0, stream>>>(hs0, P(42), P(44), P(45), xg, 64, 512, 128, 128, 0);
    k_lstm<<<1, 1024, 0, stream>>>(xg, P(43), mraw);
    k_act<<<64, 256, 0, stream>>>(mraw, m1, P(34), P(35), nullptr, 128, 1);           // gelu(LN)
    k_rowdot<<<g1(64 * 128), 256, 0, stream>>>(m1, P(32), P(33), nullptr, m2, 64, 128, 128, 128, 0);
    k_act<<<64, 256, 0, stream>>>(m2, m2, nullptr, nullptr, nullptr, 128, 3);         // gelu
    k_act<<<64, 256, 0, stream>>>(m2, mfin, P(36), P(37), nullptr, 128, 2);           // LN

    // ---- rest-pose MLP (WMMA for the big layers)
    k_rowdot<<<g1(16384 * 128), 256, 0, stream>>>(restp, P(46), P(47), nullptr, A, 16384, 128, 3, 3, 0);
    k_act<<<16384, 256, 0, stream>>>(A, A, P(48), P(49), nullptr, 128, 0);
    k_gemm16<128><<<256, 128, 0, stream>>>(A, r1Wh, P(51), Bbuf, 256);
    k_act<<<16384, 256, 0, stream>>>(Bbuf, Bbuf, P(52), P(53), nullptr, 256, 0);
    k_gemm16<256><<<256, 128, 0, stream>>>(Bbuf, r2Wh, P(55), A, 256);
    k_act<<<16384, 256, 0, stream>>>(A, A, P(56), P(57), nullptr, 256, 0);
    k_gemm16<256><<<256, 128, 0, stream>>>(A, r3Wh, P(59), restF, 128);
    k_act<<<16384, 256, 0, stream>>>(restF, restF, P(60), P(61), nullptr, 128, 0);
    k_mean<<<256, 256, 0, stream>>>(restF, restG);

    // ---- semantic MLP
    k_rowdot<<<g1(2 * 128), 256, 0, stream>>>(semv, P(62), P(63), nullptr, semA, 2, 128, 6, 6, 0);
    k_act<<<2, 256, 0, stream>>>(semA, semA, P(64), P(65), nullptr, 128, 0);
    k_rowdot<<<g1(2 * 128), 256, 0, stream>>>(semA, P(66), P(67), nullptr, semF, 2, 128, 128, 128, 0);
    k_act<<<2, 256, 0, stream>>>(semF, semF, P(68), P(69), nullptr, 128, 0);

    // ---- fusion MLP (+ residual)
    k_build_fus<<<g1(64 * 384), 256, 0, stream>>>(mfin, restG, semF, xcat);
    k_rowdot<<<g1(64 * 512), 256, 0, stream>>>(xcat, P(20), P(21), nullptr, f1, 64, 512, 384, 384, 0);
    k_act<<<64, 256, 0, stream>>>(f1, f1, P(22), P(23), nullptr, 512, 0);
    k_rowdot<<<g1(64 * 256), 256, 0, stream>>>(f1, P(24), P(25), nullptr, f2, 64, 256, 512, 512, 0);
    k_act<<<64, 256, 0, stream>>>(f2, f2, P(26), P(27), nullptr, 256, 0);
    k_rowdot<<<g1(64 * 128), 256, 0, stream>>>(f2, P(28), P(29), nullptr, fused, 64, 128, 256, 256, 0);
    k_act<<<64, 256, 0, stream>>>(fused, fused, P(30), P(31), mfin, 128, 0);          // +m residual

    // ---- decoder LSTM (2 layers) + LN
    k_rowdot<<<g1(64 * 512), 256, 0, stream>>>(fused, P(8), P(10), P(11), xg, 64, 512, 128, 128, 0);
    k_lstm<<<1, 1024, 0, stream>>>(xg, P(9), dh0);
    k_rowdot<<<g1(64 * 512), 256, 0, stream>>>(dh0, P(12), P(14), P(15), xg, 64, 512, 128, 128, 0);
    k_lstm<<<1, 1024, 0, stream>>>(xg, P(13), dh1);
    k_act<<<64, 256, 0, stream>>>(dh1, dvec, P(6), P(7), nullptr, 128, 2);

    // ---- decoder projections
    k_rowdot<<<g1(64 * 128), 256, 0, stream>>>(dvec, P(0), nullptr, nullptr, dm, 64, 128, 128, 272, 0);
    k_gemm16<128><<<256, 128, 0, stream>>>(restF, Wrh, nullptr, rvA, 128);
    k_rowdot<<<g1(8192 * 128), 256, 0, stream>>>(P(70), P(0), nullptr, nullptr, vep, 8192, 128, 16, 272, 256);
    k_combine_rv<<<g1(BB * VV * 128), 256, 0, stream>>>(rvA, vep, P(1), rv);

    // ---- fused decoder over (B, F, V) — WMMA main GEMM, 3 floats/row out
    dim3 dgrid(VV / 64, FF, BB);
    k_decoder<<<dgrid, 128, 0, stream>>>(rv, dm, W2h, P(3), P(16), P(17),
                                         P(18), P(19), P(4), P(5), (float*)d_out);
}
